// PCAProjector_87333864996936
// MI455X (gfx1250) — matrix operand order repaired and checked by reading, not compiled, not run on hardware
//
#include <hip/hip_runtime.h>
#include <hip/hip_bf16.h>

// ---------------------------------------------------------------------------
// PCAProjector for MI455X (gfx1250, wave32, WMMA).
//
//  Stage A: h = x @ W_ff + b_ff + pos_emb           (bf16 WMMA, f32 accum)
//  Stage B: per (b,n,head): center over S, cov=hh^T hh (bf16 WMMA, f32 accum),
//           Jacobi eigensolve (f32, LDS), sort eigenvalues ascending,
//           w = V @ W_point_head, partial[s] = hh_centered[s,:] . w
//  Stage C: out = sum over heads of partials + b_point
//
// Roofline: stage A moves ~157MB (memory bound at 23.3 TB/s => bf16 matrix
// path keeps compute <<< memory). h (75MB) fits in the 192MB L2 so stage B's
// re-reads are L2 hits. Eigensolve is VALU-bound but spreads over 4096 waves.
// ---------------------------------------------------------------------------

typedef __attribute__((ext_vector_type(16))) __bf16 v16bf;
typedef __attribute__((ext_vector_type(8)))  __bf16 v8bf;
typedef __attribute__((ext_vector_type(8)))  float  v8f;

#define BN     1024      // B*N = 8*128
#define SEQ    150
#define SEQP   160       // padded to 5 * 32 for WMMA K
#define DIM    128
#define NHEAD  4
#define HD     32
#define JSWEEPS 8

// strides (in bf16 elements), chosen so all fragment loads are 16B aligned
#define WT_STR 136       // 136*2 = 272B = 17*16
#define XT_STR 136
#define TT_STR 168       // 168*2 = 336B = 21*16

__device__ __forceinline__ v16bf cat8(v8bf lo, v8bf hi) {
  return __builtin_shufflevector(lo, hi, 0,1,2,3,4,5,6,7,8,9,10,11,12,13,14,15);
}

// ---------------------------------------------------------------------------
// Kernel A: per (bn, half): 80 rows x 128 cols GEMM tile, K=128.
// 256 threads = 8 waves. Wave w owns output column strip tn = w; loops over
// the 5 row tiles, so its 4 k-step B fragments are loop-invariant (compiler
// keeps them resident in VGPRs) and only A fragments stream from LDS.
// ---------------------------------------------------------------------------
__global__ __launch_bounds__(256)
void pca_gemm_ff(const float* __restrict__ x, const float* __restrict__ Wff,
                 const float* __restrict__ bff, const float* __restrict__ pos,
                 float* __restrict__ h) {
  __shared__ __align__(16) __bf16 sW[DIM * WT_STR];   // W^T: [n][k]
  __shared__ __align__(16) __bf16 sX[80  * XT_STR];   // x  : [m][k]

  const int bn    = blockIdx.x >> 1;
  const int sbase = (blockIdx.x & 1) * 80;
  const int tid   = threadIdx.x;

  // stage W_ff transposed (bf16): sW[n][k] = W[k][n]
  for (int i = tid; i < DIM * DIM; i += 256) {
    int k = i >> 7, n = i & 127;
    sW[n * WT_STR + k] = (__bf16)Wff[i];
  }
  // stage x block (bf16): sX[m][k] ; clamp rows >= SEQ (never stored)
  const float* xg = x + (size_t)bn * SEQ * DIM;
  for (int i = tid; i < 80 * DIM; i += 256) {
    int m = i >> 7, d = i & 127;
    int srow = sbase + m; if (srow > SEQ - 1) srow = SEQ - 1;
    sX[m * XT_STR + d] = (__bf16)xg[srow * DIM + d];
  }
  __syncthreads();

  const int wid   = tid >> 5;          // == tn (column tile), fixed per wave
  const int lane  = tid & 31;
  const int lhalf = lane & 15;
  const int aoff  = (lane >= 16) ? 8  : 0;   // A: K split {0-7,16-23}/{8-15,24-31}
  const int boff  = (lane >= 16) ? 8  : 0;   // B: K split {0-7,16-23}/{8-15,24-31}

  const int   col = wid * 16 + lhalf;        // global output column, invariant
  const float bc  = bff[col];
  const __bf16* pbrow = &sW[col * WT_STR + boff];

  float* hg = h + (size_t)bn * SEQ * DIM;

  for (int tm = 0; tm < 5; ++tm) {
    const int  s0   = sbase + tm * 16 + ((lane >= 16) ? 8 : 0);
    const bool full = (sbase + tm * 16 + 15) < SEQ;   // wave-uniform predicate

    // prefetch pos_emb for this tile (overlaps with the WMMA k-loop)
    float posv[8];
    if (full) {
      #pragma unroll
      for (int r = 0; r < 8; ++r) posv[r] = pos[(s0 + r) * DIM + col];
    } else {
      #pragma unroll
      for (int r = 0; r < 8; ++r)
        posv[r] = (s0 + r < SEQ) ? pos[(s0 + r) * DIM + col] : 0.f;
    }

    v8f acc = {};
    const __bf16* parow = &sX[(tm * 16 + lhalf) * XT_STR + aoff];
    #pragma unroll
    for (int kb = 0; kb < 4; ++kb) {
      const __bf16* pa = parow + kb * 32;
      const __bf16* pb = pbrow + kb * 32;
      v16bf a = cat8(*(const v8bf*)pa, *(const v8bf*)(pa + 16));
      v16bf b = cat8(*(const v8bf*)pb, *(const v8bf*)(pb + 16));
      acc = __builtin_amdgcn_wmma_f32_16x16x32_bf16(false, a, false, b,
                                                    (short)0, acc, false, false);
    }

    // epilogue: + b_ff + pos_emb, store f32. C layout: VGPR r -> M = r (+8 hi)
    if (full) {
      #pragma unroll
      for (int r = 0; r < 8; ++r)
        hg[(s0 + r) * DIM + col] = acc[r] + bc + posv[r];
    } else {
      #pragma unroll
      for (int r = 0; r < 8; ++r)
        if (s0 + r < SEQ) hg[(s0 + r) * DIM + col] = acc[r] + bc + posv[r];
    }
  }
}

// ---------------------------------------------------------------------------
// Kernel B: one wave32 per (bn, head).
// ---------------------------------------------------------------------------
__global__ __launch_bounds__(32)
void pca_head(const float* __restrict__ h, const float* __restrict__ Wp,
              float* __restrict__ part) {
  __shared__ __align__(16) __bf16 sT[HD * TT_STR];  // hh^T centered: [d][s]
  __shared__ float sA[HD * 33];                     // cov (padded stride 33)
  __shared__ float sV[HD * 33];                     // eigenvectors
  __shared__ float sC[16], sS[16];
  __shared__ int   sP[16], sQ[16], sRing[HD], sRank[HD];
  __shared__ float sWp[HD], sWv[HD];

  const int bn    = blockIdx.x >> 2;
  const int head  = blockIdx.x & 3;
  const int hbase = head * HD;
  const int lane  = threadIdx.x;

  const float* hg = h + (size_t)bn * SEQ * DIM + hbase;

  // pass 1: column mean over S (lane == d)
  float mean = 0.f;
  for (int s = 0; s < SEQ; ++s) mean += hg[s * DIM + lane];
  mean *= (1.0f / SEQ);
  // pass 2: center, convert, transpose into LDS; zero-pad K to 160
  for (int s = 0; s < SEQ; ++s)
    sT[lane * TT_STR + s] = (__bf16)(hg[s * DIM + lane] - mean);
  for (int s = SEQ; s < SEQP; ++s)
    sT[lane * TT_STR + s] = (__bf16)0.f;

  sRing[lane] = lane;
  for (int c = 0; c < HD; ++c) sV[lane * 33 + c] = (lane == c) ? 1.f : 0.f;
  sWp[lane] = Wp[hbase + lane];
  __syncthreads();

  // cov = hh^T hh via WMMA (scale 1/(S-1) omitted: eigenvectors unaffected)
  const int lhalf = lane & 15;
  const int aoff  = (lane >= 16) ? 8  : 0;
  const int boff  = (lane >= 16) ? 8  : 0;
  #pragma unroll
  for (int tm = 0; tm < 2; ++tm) {
    #pragma unroll
    for (int tn = 0; tn < 2; ++tn) {
      v8f acc = {};
      const __bf16* parow = &sT[(tm * 16 + lhalf) * TT_STR + aoff];
      const __bf16* pbrow = &sT[(tn * 16 + lhalf) * TT_STR + boff];
      #pragma unroll
      for (int kb = 0; kb < 5; ++kb) {
        const __bf16* pa = parow + kb * 32;
        const __bf16* pb = pbrow + kb * 32;
        v16bf a = cat8(*(const v8bf*)pa, *(const v8bf*)(pa + 16));
        v16bf b = cat8(*(const v8bf*)pb, *(const v8bf*)(pb + 16));
        acc = __builtin_amdgcn_wmma_f32_16x16x32_bf16(false, a, false, b,
                                                      (short)0, acc, false, false);
      }
      const int row0 = tm * 16 + ((lane >= 16) ? 8 : 0);
      const int col  = tn * 16 + lhalf;
      #pragma unroll
      for (int r = 0; r < 8; ++r) sA[(row0 + r) * 33 + col] = acc[r];
    }
  }
  __syncthreads();

  // cyclic Jacobi, round-robin tournament pairing: 16 disjoint rotations/round
  for (int sweep = 0; sweep < JSWEEPS; ++sweep) {
    for (int rr = 0; rr < HD - 1; ++rr) {
      if (lane < 16) {
        int p = sRing[lane], q = sRing[31 - lane];
        float app = sA[p * 33 + p], aqq = sA[q * 33 + q], apq = sA[p * 33 + q];
        float c = 1.f, s = 0.f;
        if (fabsf(apq) > 1e-12f) {
          float tau = (aqq - app) / (2.f * apq);
          float tt  = ((tau >= 0.f) ? 1.f : -1.f) /
                      (fabsf(tau) + sqrtf(1.f + tau * tau));
          c = 1.f / sqrtf(1.f + tt * tt);
          s = tt * c;
        }
        sC[lane] = c; sS[lane] = s; sP[lane] = p; sQ[lane] = q;
      }
      __syncthreads();
      // row rotations (A' = J^T A): rows disjoint across j
      for (int j = 0; j < 16; ++j) {
        int p = sP[j], q = sQ[j]; float c = sC[j], s = sS[j];
        float ap = sA[p * 33 + lane], aq = sA[q * 33 + lane];
        sA[p * 33 + lane] = c * ap - s * aq;
        sA[q * 33 + lane] = s * ap + c * aq;
      }
      __syncthreads();
      // column rotations (A'' = A' J) and V' = V J
      for (int j = 0; j < 16; ++j) {
        int p = sP[j], q = sQ[j]; float c = sC[j], s = sS[j];
        float ap = sA[lane * 33 + p], aq = sA[lane * 33 + q];
        sA[lane * 33 + p] = c * ap - s * aq;
        sA[lane * 33 + q] = s * ap + c * aq;
        float vp = sV[lane * 33 + p], vq = sV[lane * 33 + q];
        sV[lane * 33 + p] = c * vp - s * vq;
        sV[lane * 33 + q] = s * vp + c * vq;
      }
      __syncthreads();
      // rotate ring positions 1..31 (position 0 fixed)
      int nv = 0;
      if (lane > 0) nv = sRing[(lane == 1) ? 31 : lane - 1];
      __syncthreads();
      if (lane > 0) sRing[lane] = nv;
      __syncthreads();
    }
  }

  // rank eigenvalues ascending (eigh order), pair with W_point entries
  {
    float lam = sA[lane * 33 + lane];
    int r = 0;
    for (int j = 0; j < HD; ++j) {
      float lj = sA[j * 33 + j];
      r += (lj < lam) || (lj == lam && j < lane);
    }
    sRank[lane] = r;
  }
  __syncthreads();
  {
    float w = 0.f;
    for (int c = 0; c < HD; ++c) w += sV[lane * 33 + c] * sWp[sRank[c]];
    sWv[lane] = w;
  }
  __syncthreads();

  // partial[s] = centered_hh[s,:] . w
  float* pg = part + ((size_t)bn * SEQ) * NHEAD + head;
  for (int i = 0; i < 5; ++i) {
    int s = lane + 32 * i;
    if (s < SEQ) {
      float acc = 0.f;
      for (int d = 0; d < HD; ++d)
        acc += (float)sT[d * TT_STR + s] * sWv[d];
      pg[(size_t)s * NHEAD] = acc;
    }
  }
}

// ---------------------------------------------------------------------------
// Kernel C: sum 4 head partials + b_point
// ---------------------------------------------------------------------------
__global__ __launch_bounds__(256)
void pca_combine(const float* __restrict__ part, const float* __restrict__ bp,
                 float* __restrict__ out, int n) {
  int i = blockIdx.x * 256 + threadIdx.x;
  if (i < n) {
    const float* p = part + (size_t)i * NHEAD;
    out[i] = p[0] + p[1] + p[2] + p[3] + bp[0];
  }
}

// ---------------------------------------------------------------------------
extern "C" void kernel_launch(void* const* d_in, const int* in_sizes, int n_in,
                              void* d_out, int out_size, void* d_ws, size_t ws_size,
                              hipStream_t stream) {
  const float* x   = (const float*)d_in[0];
  const float* Wff = (const float*)d_in[1];
  const float* bff = (const float*)d_in[2];
  const float* pos = (const float*)d_in[3];
  const float* Wp  = (const float*)d_in[4];
  const float* bp  = (const float*)d_in[5];
  float* out = (float*)d_out;

  // workspace: h [BN,SEQ,DIM] f32, then part [BN,SEQ,NHEAD] f32
  float* h    = (float*)d_ws;
  float* part = h + (size_t)BN * SEQ * DIM;

  pca_gemm_ff<<<BN * 2, 256, 0, stream>>>(x, Wff, bff, pos, h);
  pca_head<<<BN * NHEAD, 32, 0, stream>>>(h, Wp, part);
  const int n = BN * SEQ;
  pca_combine<<<(n + 255) / 256, 256, 0, stream>>>(part, bp, out, n);
}
